// MatchaModel_51866025066595
// MI455X (gfx1250) — matrix operand order repaired and checked by reading, not compiled, not run on hardware
//
#include <hip/hip_runtime.h>
#include <hip/hip_bf16.h>
#include <math.h>

// ---------------------------------------------------------------------------
// MatchaModel (DiT-like 6-block transformer) for gfx1250 / MI455X.
// GEMMs: v_wmma_f32_16x16x32_bf16, tiles staged into LDS with
// global_load_async_to_lds_b128 (ASYNCcnt path). f32 residual/softmax path.
// ---------------------------------------------------------------------------

typedef __attribute__((ext_vector_type(16))) __bf16 v16bf;
typedef __attribute__((ext_vector_type(8)))  __bf16 v8bf;
typedef __attribute__((ext_vector_type(8)))  float  v8f;

#define LNUM 6
#define BB   4
#define NN   512
#define CC   512
#define HH   8
#define DD   64
#define MLPD 2048
#define SIXC (6*CC)

__device__ __forceinline__ v16bf cat8(v8bf lo, v8bf hi) {
  return __builtin_shufflevector(lo, hi, 0,1,2,3,4,5,6,7,8,9,10,11,12,13,14,15);
}

__device__ __forceinline__ v8f wmma_bf16(v16bf a, v16bf b, v8f c) {
  // D = A(16x32) x B(32x16) + C, f32 accumulate
  return __builtin_amdgcn_wmma_f32_16x16x32_bf16(false, a, false, b, (short)0, c,
                                                 false, false);
}

// Async global->LDS 16B copy (CDNA5, tracked with ASYNCcnt).
__device__ __forceinline__ void async_ld_b128(const void* gptr, void* lptr) {
  unsigned lds_off = (unsigned)(unsigned long long)lptr;  // low 32b = LDS byte addr
  asm volatile("global_load_async_to_lds_b128 %0, %1, off"
               :: "v"(lds_off), "v"(gptr) : "memory");
}
__device__ __forceinline__ void wait_async0() {
  asm volatile("s_wait_asynccnt 0x0" ::: "memory");
}

// ---------------------------------------------------------------------------
// adaLN: ada[l,b,:] = silu(time_emb[b]) @ wada[l] + bada[l]   (all blocks)
// ---------------------------------------------------------------------------
__global__ __launch_bounds__(256)
void ada_kernel(const float* __restrict__ time_emb,
                const float* __restrict__ wada,
                const float* __restrict__ bada,
                float* __restrict__ ada) {
  const int b = blockIdx.x, l = blockIdx.y, tid = threadIdx.x;
  __shared__ float t[CC];
  for (int c = tid; c < CC; c += 256) {
    float x = time_emb[b*CC + c];
    t[c] = x / (1.0f + __expf(-x));
  }
  __syncthreads();
  const float* wl = wada + (size_t)l*CC*SIXC;
  for (int j = tid; j < SIXC; j += 256) {
    float s = bada[l*SIXC + j];
    for (int c = 0; c < CC; ++c) s = fmaf(t[c], wl[(size_t)c*SIXC + j], s);
    ada[((size_t)(l*BB + b))*SIXC + j] = s;
  }
}

// ---------------------------------------------------------------------------
// Weight convert + transpose: w[K,N] f32 -> wt[N,K] bf16
// ---------------------------------------------------------------------------
__global__ __launch_bounds__(256)
void wt_kernel(const float* __restrict__ w, __bf16* __restrict__ wt,
               int kshift, int N) {
  size_t idx = (size_t)blockIdx.x*256 + threadIdx.x;   // over N*K (wt order n*K+k)
  int K = 1 << kshift;
  int k = (int)(idx & (size_t)(K - 1));
  int n = (int)(idx >> kshift);
  wt[idx] = (__bf16)w[(size_t)k*N + n];
}

// ---------------------------------------------------------------------------
// LayerNorm (no affine, eps=1e-6) + modulate -> bf16. One 256-thread block/row.
// ---------------------------------------------------------------------------
__global__ __launch_bounds__(256)
void ln_mod_kernel(const float* __restrict__ x,
                   const float* __restrict__ adal,   // ada + l*B*6C
                   __bf16* __restrict__ out,
                   int shiftSel, int scaleSel) {
  const int row = blockIdx.x;        // b*N + n
  const int b   = row >> 9;          // N = 512
  const int tid = threadIdx.x;
  __shared__ float red[256];
  float x0 = x[(size_t)row*CC + tid];
  float x1 = x[(size_t)row*CC + tid + 256];
  red[tid] = x0 + x1;
  __syncthreads();
  for (int s = 128; s > 0; s >>= 1) { if (tid < s) red[tid] += red[tid+s]; __syncthreads(); }
  float mu = red[0] * (1.0f/CC);
  __syncthreads();
  float d0 = x0 - mu, d1 = x1 - mu;
  red[tid] = d0*d0 + d1*d1;
  __syncthreads();
  for (int s = 128; s > 0; s >>= 1) { if (tid < s) red[tid] += red[tid+s]; __syncthreads(); }
  float rstd = rsqrtf(red[0]*(1.0f/CC) + 1e-6f);
  const float* ab = adal + (size_t)b*SIXC;
  float sh0 = ab[shiftSel*CC + tid],       sc0 = ab[scaleSel*CC + tid];
  float sh1 = ab[shiftSel*CC + tid + 256], sc1 = ab[scaleSel*CC + tid + 256];
  out[(size_t)row*CC + tid]       = (__bf16)((d0*rstd)*(1.0f+sc0) + sh0);
  out[(size_t)row*CC + tid + 256] = (__bf16)((d1*rstd)*(1.0f+sc1) + sh1);
}

// ---------------------------------------------------------------------------
// Tiled WMMA GEMM: C[M,N] = A[M,K](bf16,row-major) @ Bt[N,K](bf16, pre-transposed)
// 128x128 tile / workgroup, 8 waves (2x4), each wave 64x32 = 4x2 WMMA tiles.
// Both tiles staged via global_load_async_to_lds_b128.
// EPI 0: store bf16.   EPI 1: +bias, tanh-GELU, store bf16.
// EPI 2: Cres[m,n] += gate[b,n] * (acc + bias[n])   (residual update, N==512)
// ---------------------------------------------------------------------------
template<int EPI>
__global__ __launch_bounds__(256)
void gemm_kernel(const __bf16* __restrict__ A,
                 const __bf16* __restrict__ Bt,
                 const float* __restrict__ bias,
                 const float* __restrict__ gate,   // pre-offset by sel*C; stride 6C/batch
                 float* __restrict__ Cres,
                 __bf16* __restrict__ Cbf,
                 int M, int N, int K) {
  __shared__ __align__(16) __bf16 As[128][32];   // [m][k]
  __shared__ __align__(16) __bf16 Bs[128][32];   // [n][k]
  const int tid  = threadIdx.x;
  const int lane = tid & 31;
  const int wave = tid >> 5;
  const int bm = blockIdx.y * 128;
  const int bn = blockIdx.x * 128;
  const int wm = (wave >> 2) * 64;
  const int wn = (wave & 3) * 32;
  const int hf = lane >> 4;      // half-wave
  const int lr = lane & 15;

  v8f acc[4][2];
  #pragma unroll
  for (int i = 0; i < 4; ++i)
    #pragma unroll
    for (int j = 0; j < 2; ++j)
      #pragma unroll
      for (int e = 0; e < 8; ++e) acc[i][j][e] = 0.0f;

  // per-thread tile-fill coordinates (8 bf16 = 16B per pass, 2 passes)
  const int r0 = (tid*8) >> 5,          c0 = (tid*8) & 31;
  const int r1 = (tid*8 + 2048) >> 5,   c1 = (tid*8 + 2048) & 31;

  for (int k0 = 0; k0 < K; k0 += 32) {
    async_ld_b128(A  + (size_t)(bm + r0)*K + k0 + c0, &As[r0][c0]);
    async_ld_b128(A  + (size_t)(bm + r1)*K + k0 + c1, &As[r1][c1]);
    async_ld_b128(Bt + (size_t)(bn + r0)*K + k0 + c0, &Bs[r0][c0]);
    async_ld_b128(Bt + (size_t)(bn + r1)*K + k0 + c1, &Bs[r1][c1]);
    if (k0 + 32 < K) {   // prefetch next K-tile
      __builtin_prefetch(A  + (size_t)(bm + r0)*K + k0 + 32 + c0, 0, 1);
      __builtin_prefetch(Bt + (size_t)(bn + r0)*K + k0 + 32 + c0, 0, 1);
    }
    wait_async0();
    __syncthreads();

    // Fragments (A 16x32 layout: K-groups {8h..8h+7, 8h+16..8h+23})
    v16bf af[4], bfr[2];
    #pragma unroll
    for (int i = 0; i < 4; ++i) {
      const __bf16* p = &As[wm + i*16 + lr][8*hf];
      af[i] = cat8(*(const v8bf*)p, *(const v8bf*)(p + 16));
    }
    #pragma unroll
    for (int j = 0; j < 2; ++j) {
      const __bf16* p = &Bs[wn + j*16 + lr][8*hf];
      bfr[j] = cat8(*(const v8bf*)p, *(const v8bf*)(p + 16));
    }
    #pragma unroll
    for (int i = 0; i < 4; ++i)
      #pragma unroll
      for (int j = 0; j < 2; ++j)
        acc[i][j] = wmma_bf16(af[i], bfr[j], acc[i][j]);
    __syncthreads();
  }

  // Epilogue: C/D frag layout -> m = tile_m + r + 8*hf, n = tile_n + lr
  #pragma unroll
  for (int i = 0; i < 4; ++i) {
    #pragma unroll
    for (int j = 0; j < 2; ++j) {
      #pragma unroll
      for (int r = 0; r < 8; ++r) {
        int m = bm + wm + i*16 + hf*8 + r;
        int n = bn + wn + j*16 + lr;
        float v = acc[i][j][r];
        if (EPI == 0) {
          Cbf[(size_t)m*N + n] = (__bf16)v;
        } else if (EPI == 1) {
          v += bias[n];
          float u = 0.7978845608028654f * (v + 0.044715f*v*v*v);
          v = 0.5f * v * (1.0f + tanhf(u));
          Cbf[(size_t)m*N + n] = (__bf16)v;
        } else {
          v += bias[n];
          float g = gate[(size_t)(m >> 9)*SIXC + n];   // batch = m/512
          Cres[(size_t)m*N + n] += g * v;
        }
      }
    }
  }
}

// ---------------------------------------------------------------------------
// V transpose: (b, key, h, d) -> vt[(b,h,d), key]  (bf16)
// ---------------------------------------------------------------------------
__global__ __launch_bounds__(256)
void vtrans_kernel(const __bf16* __restrict__ v, __bf16* __restrict__ vt) {
  size_t idx = (size_t)blockIdx.x*256 + threadIdx.x;   // over B*N*C
  int d   = (int)(idx & 63);
  int h   = (int)((idx >> 6) & 7);
  int key = (int)((idx >> 9) & 511);
  int b   = (int)(idx >> 18);
  vt[((size_t)(b*HH + h)*DD + d)*NN + key] = v[idx];
}

// ---------------------------------------------------------------------------
// Fused attention: per (b, h, 32 query rows). 2 waves, 16 rows each.
// logits = qk*scale + where(mask,-inf,pair); pair <- logits (in place);
// softmax in LDS; o = probs @ V  (all matmuls via WMMA bf16).
// ---------------------------------------------------------------------------
__global__ __launch_bounds__(64)
void attn_kernel(const __bf16* __restrict__ q,
                 const __bf16* __restrict__ k,
                 const __bf16* __restrict__ vt,
                 float* __restrict__ pair,
                 const unsigned char* __restrict__ mask,
                 __bf16* __restrict__ o) {
  __shared__ float sc[32][NN];                 // 64 KB
  const int tid  = threadIdx.x;
  const int lane = tid & 31;
  const int wave = tid >> 5;
  const int hf = lane >> 4, lr = lane & 15;
  const int h  = blockIdx.y;
  const int b  = blockIdx.z;
  const int qb = blockIdx.x*32 + wave*16;      // wave's 16 query rows
  const float scale_ = 0.125f;                 // D^-0.5, D=64

  // Q fragments: K(=head dim 64) in two 32-chunks; contiguous in global.
  v16bf qf[2];
  {
    const __bf16* qp = q + (size_t)(b*NN + qb + lr)*CC + h*DD;
    #pragma unroll
    for (int kk = 0; kk < 2; ++kk) {
      const __bf16* p = qp + kk*32 + hf*8;
      qf[kk] = cat8(*(const v8bf*)p, *(const v8bf*)(p + 16));
    }
  }

  // ---- QK^T over 32 key tiles of 16 ----
  for (int kt = 0; kt < 32; ++kt) {
    int key = kt*16 + lr;
    const __bf16* kp = k + (size_t)(b*NN + key)*CC + h*DD;
    v8f a;
    #pragma unroll
    for (int e = 0; e < 8; ++e) a[e] = 0.0f;
    #pragma unroll
    for (int kk = 0; kk < 2; ++kk) {
      const __bf16* p = kp + kk*32 + hf*8;
      v16bf kf = cat8(*(const v8bf*)p, *(const v8bf*)(p + 16));
      a = wmma_bf16(qf[kk], kf, a);
    }
    const bool mk = (mask[b*NN + key] != 0);
    #pragma unroll
    for (int r = 0; r < 8; ++r) {
      int qrow = qb + hf*8 + r;
      size_t pidx = ((size_t)(b*HH + h)*NN + qrow)*NN + key;
      float pv = pair[pidx];                      // unconditional load
      pv = mk ? -INFINITY : pv;                   // v_cndmask, no exec branch
      float logit = fmaf(a[r], scale_, pv);
      pair[pidx] = logit;                         // pair_emb residual update
      sc[wave*16 + hf*8 + r][key] = logit;
    }
  }
  __syncthreads();

  // ---- softmax, one lane per row ----
  if (tid < 32) {
    float m = -INFINITY;
    for (int j = 0; j < NN; ++j) m = fmaxf(m, sc[tid][j]);
    float s = 0.0f;
    for (int j = 0; j < NN; ++j) { float e = __expf(sc[tid][j] - m); sc[tid][j] = e; s += e; }
    float inv = 1.0f / s;
    for (int j = 0; j < NN; ++j) sc[tid][j] *= inv;
  }
  __syncthreads();

  // ---- probs @ V ----
  v8f oacc[4];
  #pragma unroll
  for (int dt = 0; dt < 4; ++dt)
    #pragma unroll
    for (int e = 0; e < 8; ++e) oacc[dt][e] = 0.0f;

  for (int kk = 0; kk < NN; kk += 32) {
    v16bf af;
    const float* srow = &sc[wave*16 + lr][kk + hf*8];
    #pragma unroll
    for (int e = 0; e < 8; ++e) { af[e] = (__bf16)srow[e]; af[e+8] = (__bf16)srow[e+16]; }
    #pragma unroll
    for (int dt = 0; dt < 4; ++dt) {
      const __bf16* p = vt + ((size_t)(b*HH + h)*DD + dt*16 + lr)*NN + kk + hf*8;
      v16bf vf = cat8(*(const v8bf*)p, *(const v8bf*)(p + 16));
      oacc[dt] = wmma_bf16(af, vf, oacc[dt]);
    }
  }
  #pragma unroll
  for (int dt = 0; dt < 4; ++dt)
    #pragma unroll
    for (int r = 0; r < 8; ++r) {
      int qrow = qb + hf*8 + r;
      o[(size_t)(b*NN + qrow)*CC + h*DD + dt*16 + lr] = (__bf16)oacc[dt][r];
    }
}

// ---------------------------------------------------------------------------
extern "C" void kernel_launch(void* const* d_in, const int* in_sizes, int n_in,
                              void* d_out, int out_size, void* d_ws, size_t ws_size,
                              hipStream_t stream) {
  (void)in_sizes; (void)n_in; (void)out_size; (void)ws_size;
  const float* x_in    = (const float*)d_in[0];
  const float* temb    = (const float*)d_in[1];
  const float* pair_in = (const float*)d_in[2];
  const unsigned char* mask = (const unsigned char*)d_in[3];
  const float* wq = (const float*)d_in[4];
  const float* wk = (const float*)d_in[5];
  const float* wv = (const float*)d_in[6];
  const float* wo = (const float*)d_in[7];
  const float* bo = (const float*)d_in[8];
  const float* w1 = (const float*)d_in[9];
  const float* b1 = (const float*)d_in[10];
  const float* w2 = (const float*)d_in[11];
  const float* b2 = (const float*)d_in[12];
  const float* wada = (const float*)d_in[13];
  const float* bada = (const float*)d_in[14];

  float* xo    = (float*)d_out;                 // (B,N,C)
  float* pairo = xo + (size_t)BB*NN*CC;         // (B,H,N,N)

  char* w = (char*)d_ws;
  float*  ada = (float*) (w + 0x0000000);       // L*B*6C f32 = 0x48000 B
  __bf16* mx  = (__bf16*)(w + 0x0048000);       // (B*N, C)   bf16, 2 MB
  __bf16* qb  = (__bf16*)(w + 0x0248000);
  __bf16* kb  = (__bf16*)(w + 0x0448000);
  __bf16* vb  = (__bf16*)(w + 0x0648000);
  __bf16* vt  = (__bf16*)(w + 0x0848000);       // (B,H,D,N)  bf16
  __bf16* ob  = (__bf16*)(w + 0x0A48000);       // attn out   bf16
  __bf16* h1  = (__bf16*)(w + 0x0C48000);       // (B*N, MLP) bf16, 8 MB
  __bf16* wqT = (__bf16*)(w + 0x1448000);       // (C,C)      bf16, 512 KB each
  __bf16* wkT = (__bf16*)(w + 0x14C8000);
  __bf16* wvT = (__bf16*)(w + 0x1548000);
  __bf16* woT = (__bf16*)(w + 0x15C8000);
  __bf16* w1T = (__bf16*)(w + 0x1648000);       // (MLP,C)    bf16, 2 MB
  __bf16* w2T = (__bf16*)(w + 0x1848000);       // (C,MLP)    bf16, 2 MB

  hipMemcpyAsync(xo, x_in, (size_t)BB*NN*CC*sizeof(float),
                 hipMemcpyDeviceToDevice, stream);
  hipMemcpyAsync(pairo, pair_in, (size_t)BB*HH*NN*NN*sizeof(float),
                 hipMemcpyDeviceToDevice, stream);

  ada_kernel<<<dim3(BB, LNUM), 256, 0, stream>>>(temb, wada, bada, ada);

  dim3 g512(CC/128, (BB*NN)/128);     // GEMMs with N=512
  dim3 gmlp(MLPD/128, (BB*NN)/128);   // fc1 with N=2048

  for (int l = 0; l < LNUM; ++l) {
    const float* adal = ada + (size_t)l*BB*SIXC;
    // per-layer weight convert+transpose to bf16 [N][K]
    wt_kernel<<<(CC*CC)/256,   256, 0, stream>>>(wq + (size_t)l*CC*CC,  wqT, 9,  CC);
    wt_kernel<<<(CC*CC)/256,   256, 0, stream>>>(wk + (size_t)l*CC*CC,  wkT, 9,  CC);
    wt_kernel<<<(CC*CC)/256,   256, 0, stream>>>(wv + (size_t)l*CC*CC,  wvT, 9,  CC);
    wt_kernel<<<(CC*CC)/256,   256, 0, stream>>>(wo + (size_t)l*CC*CC,  woT, 9,  CC);
    wt_kernel<<<(CC*MLPD)/256, 256, 0, stream>>>(w1 + (size_t)l*CC*MLPD, w1T, 9,  MLPD);
    wt_kernel<<<(CC*MLPD)/256, 256, 0, stream>>>(w2 + (size_t)l*MLPD*CC, w2T, 11, CC);

    // MSA branch
    ln_mod_kernel<<<BB*NN, 256, 0, stream>>>(xo, adal, mx, 0, 1);
    gemm_kernel<0><<<g512, 256, 0, stream>>>(mx, wqT, nullptr, nullptr, nullptr, qb, BB*NN, CC, CC);
    gemm_kernel<0><<<g512, 256, 0, stream>>>(mx, wkT, nullptr, nullptr, nullptr, kb, BB*NN, CC, CC);
    gemm_kernel<0><<<g512, 256, 0, stream>>>(mx, wvT, nullptr, nullptr, nullptr, vb, BB*NN, CC, CC);
    vtrans_kernel<<<(BB*NN*CC)/256, 256, 0, stream>>>(vb, vt);
    attn_kernel<<<dim3(NN/32, HH, BB), 64, 0, stream>>>(qb, kb, vt, pairo, mask, ob);
    gemm_kernel<2><<<g512, 256, 0, stream>>>(ob, woT, bo + (size_t)l*CC,
                                             adal + 2*CC, xo, nullptr, BB*NN, CC, CC);
    // MLP branch
    ln_mod_kernel<<<BB*NN, 256, 0, stream>>>(xo, adal, mx, 3, 4);
    gemm_kernel<1><<<gmlp, 256, 0, stream>>>(mx, w1T, b1 + (size_t)l*MLPD,
                                             nullptr, nullptr, h1, BB*NN, MLPD, CC);
    gemm_kernel<2><<<g512, 256, 0, stream>>>(h1, w2T, b2 + (size_t)l*CC,
                                             adal + 5*CC, xo, nullptr, BB*NN, CC, MLPD);
  }
}
